// CL_DTA_11639361372772
// MI455X (gfx1250) — matrix operand order repaired and checked by reading, not compiled, not run on hardware
//
#include <hip/hip_runtime.h>

typedef __attribute__((ext_vector_type(16))) __bf16 v16bf;
typedef __attribute__((ext_vector_type(2)))  __bf16 v2bf;
typedef __attribute__((ext_vector_type(8)))  float  v8f;

// ---------- bf16 helpers ----------
__device__ __forceinline__ unsigned short bfbits(float f) {
    unsigned u = __float_as_uint(f);
    return (unsigned short)((u + 0x7FFFu + ((u >> 16) & 1u)) >> 16);  // RNE
}
__device__ __forceinline__ unsigned pack2(float lo, float hi) {
#if __has_builtin(__builtin_amdgcn_cvt_pk_bf16_f32)
    v2bf p = __builtin_amdgcn_cvt_pk_bf16_f32(lo, hi);
    unsigned u;
    __builtin_memcpy(&u, &p, 4);
    return u;
#else
    return ((unsigned)bfbits(hi) << 16) | (unsigned)bfbits(lo);
#endif
}

__device__ __forceinline__ void atomicMaxF(float* a, float v) {
    if (v >= 0.0f) atomicMax((int*)a, __float_as_int(v));
    else           atomicMin((unsigned int*)a, __float_as_uint(v));
}

// ---------- LDS-staged WMMA GEMM: C[M,N] = A[M,K] @ W[N,K]^T + bias ----------
// Requires K % 32 == 0. 256 threads = 8 waves; 64x64 tile; each wave -> two 16x16 subtiles.
// OOB tile rows are CLAMPED (not zeroed): an OOB A-row only feeds C-rows we never
// store, and an OOB W-row only feeds C-cols we never store, so garbage is harmless
// and the stage loads stay branch-free.
#define BM 64
#define BN 64
#define BK 32
__global__ __launch_bounds__(256) void gemm_tile(
        const float* __restrict__ A, const float* __restrict__ W,
        const float* __restrict__ bias, float* __restrict__ C,
        int M, int K, int N, int act) {
    __shared__ alignas(16) unsigned short As[BM * BK];  // [m][k] bf16
    __shared__ alignas(16) unsigned short Bs[BN * BK];  // [n][k] bf16

    const int tid  = threadIdx.x;
    const int lane = tid & 31;
    const int w    = tid >> 5;          // wave 0..7
    const int wm   = w & 3;             // subtile row 0..3
    const int wn   = w >> 2;            // subtile col-pair 0..1
    const int mi   = lane & 15;
    const int hi   = lane >> 4;
    const int m0   = blockIdx.x * BM;
    const int n0   = blockIdx.y * BN;

    // cooperative load mapping: each thread moves 8 floats of A and 8 of B per K-step
    const int ar = tid >> 2;            // tile row (0..63)
    const int ak = (tid & 3) * 8;       // k-offset (0,8,16,24)

    int gm = m0 + ar; if (gm >= M) gm = M - 1;   // clamp (see note above)
    int gn = n0 + ar; if (gn >= N) gn = N - 1;
    const float* pa = A + (size_t)gm * K + ak;
    const float* pw = W + (size_t)gn * K + ak;

    v8f acc0 = {}, acc1 = {};

    for (int k0 = 0; k0 < K; k0 += BK) {
        // ---- stage A tile ----
        {
            float4 f0 = *(const float4*)(pa + k0);
            float4 f1 = *(const float4*)(pa + k0 + 4);
            uint4 pk;
            pk.x = pack2(f0.x, f0.y); pk.y = pack2(f0.z, f0.w);
            pk.z = pack2(f1.x, f1.y); pk.w = pack2(f1.z, f1.w);
            *(uint4*)&As[ar * BK + ak] = pk;
        }
        // ---- stage B tile (B[k][n] = W[n][k], stored [n][k]) ----
        {
            float4 f0 = *(const float4*)(pw + k0);
            float4 f1 = *(const float4*)(pw + k0 + 4);
            uint4 pk;
            pk.x = pack2(f0.x, f0.y); pk.y = pack2(f0.z, f0.w);
            pk.z = pack2(f1.x, f1.y); pk.w = pack2(f1.z, f1.w);
            *(uint4*)&Bs[ar * BK + ak] = pk;
        }
        __syncthreads();

        // ---- fragments straight from LDS (contiguous b128 loads) ----
        const unsigned short* Arow = &As[(wm * 16 + mi) * BK];
        uint4 alo = *(const uint4*)&Arow[hi * 16];       // K = hi*16 + 0..7
        uint4 ahi = *(const uint4*)&Arow[hi * 16 + 8];   // K = hi*16 + 8..15
        v16bf a;
        __builtin_memcpy(&a, &alo, 16);
        __builtin_memcpy((char*)&a + 16, &ahi, 16);

        const unsigned short* B0 = &Bs[(wn * 32 + mi) * BK];
        uint4 b0lo = *(const uint4*)&B0[hi * 16];        // K = hi*16 + 0..7
        uint4 b0hi = *(const uint4*)&B0[hi * 16 + 8];    // K = hi*16 + 8..15
        v16bf b0;
        __builtin_memcpy(&b0, &b0lo, 16);
        __builtin_memcpy((char*)&b0 + 16, &b0hi, 16);

        const unsigned short* B1 = &Bs[(wn * 32 + 16 + mi) * BK];
        uint4 b1lo = *(const uint4*)&B1[hi * 16];
        uint4 b1hi = *(const uint4*)&B1[hi * 16 + 8];
        v16bf b1;
        __builtin_memcpy(&b1, &b1lo, 16);
        __builtin_memcpy((char*)&b1 + 16, &b1hi, 16);

        acc0 = __builtin_amdgcn_wmma_f32_16x16x32_bf16(false, a, false, b0,
                                                       (short)0, acc0, false, false);
        acc1 = __builtin_amdgcn_wmma_f32_16x16x32_bf16(false, a, false, b1,
                                                       (short)0, acc1, false, false);
        __syncthreads();
    }

    // ---- epilogue ----
#pragma unroll
    for (int s = 0; s < 2; ++s) {
        v8f acc = s ? acc1 : acc0;
        int nn = n0 + wn * 32 + s * 16 + mi;
        if (nn < N) {
            float bv = bias[nn];
#pragma unroll
            for (int i = 0; i < 8; ++i) {
                int mm = m0 + wm * 16 + i + hi * 8;
                if (mm < M) {
                    float o = acc[i] + bv;
                    if (act == 1) o = fmaxf(o, 0.0f);
                    C[(size_t)mm * N + nn] = o;
                }
            }
        }
    }
}

// ---------- pad rows with zeros: [rows, sk] -> [rows, dk] ----------
__global__ void pad_rows_k(const float* __restrict__ src, float* __restrict__ dst,
                           int sk, int dk) {
    int r = blockIdx.x, t = threadIdx.x;
    for (int c = t; c < dk; c += blockDim.x)
        dst[(size_t)r * dk + c] = (c < sk) ? src[(size_t)r * sk + c] : 0.0f;
}

// ---------- N=1 head: out[row] = dot(A[row,:], w) + b ----------
__global__ void dot_head_k(const float* __restrict__ A, const float* __restrict__ w,
                           const float* __restrict__ b, float* __restrict__ out, int K) {
    int row = blockIdx.x, t = threadIdx.x;   // block 128
    __shared__ float red[128];
    float s = 0.0f;
    for (int c = t; c < K; c += 128) s += A[(size_t)row * K + c] * w[c];
    red[t] = s; __syncthreads();
    for (int o = 64; o > 0; o >>= 1) { if (t < o) red[t] += red[t + o]; __syncthreads(); }
    if (t == 0) out[row] = red[0] + b[0];
}

// ---------- fill ----------
__global__ void fill_k(float* p, float v, size_t n) {
    size_t i = (size_t)blockIdx.x * blockDim.x + threadIdx.x;
    if (i < n) p[i] = v;
}

// ---------- GATv2 edge kernels (E real edges + N self loops) ----------
__global__ void gat_alpha_k(const int* __restrict__ ei, const float* __restrict__ xl,
                            const float* __restrict__ xr, const float* __restrict__ att,
                            float* __restrict__ alpha, int E, int N) {
    int e = blockIdx.x * blockDim.x + threadIdx.x;
    if (e >= E + N) return;
    int s = (e < E) ? ei[e]     : (e - E);
    int d = (e < E) ? ei[E + e] : (e - E);
    const float* pl = xl + (size_t)s * 512;
    const float* pr = xr + (size_t)d * 512;
    for (int h = 0; h < 4; ++h) {
        float acc = 0.0f;
        for (int c = 0; c < 128; ++c) {
            float m = pl[h * 128 + c] + pr[h * 128 + c];
            m = (m > 0.0f) ? m : 0.2f * m;               // leaky_relu(0.2)
            acc += m * att[h * 128 + c];
        }
        alpha[(size_t)e * 4 + h] = acc;
    }
}

__global__ void gat_amax_k(const int* __restrict__ ei, const float* __restrict__ alpha,
                           float* __restrict__ amax, int E, int N) {
    int e = blockIdx.x * blockDim.x + threadIdx.x;
    if (e >= E + N) return;
    int d = (e < E) ? ei[E + e] : (e - E);
    for (int h = 0; h < 4; ++h)
        atomicMaxF(&amax[(size_t)d * 4 + h], alpha[(size_t)e * 4 + h]);
}

__global__ void gat_expden_k(const int* __restrict__ ei, float* __restrict__ alpha,
                             const float* __restrict__ amax, float* __restrict__ den,
                             int E, int N) {
    int e = blockIdx.x * blockDim.x + threadIdx.x;
    if (e >= E + N) return;
    int d = (e < E) ? ei[E + e] : (e - E);
    for (int h = 0; h < 4; ++h) {
        float ex = expf(alpha[(size_t)e * 4 + h] - amax[(size_t)d * 4 + h]);
        alpha[(size_t)e * 4 + h] = ex;
        atomicAdd(&den[(size_t)d * 4 + h], ex);
    }
}

__global__ void gat_scatter_k(const int* __restrict__ ei, const float* __restrict__ ex,
                              const float* __restrict__ den, const float* __restrict__ xl,
                              float* __restrict__ hout, int E, int N) {
    int e = blockIdx.x * blockDim.x + threadIdx.x;
    if (e >= E + N) return;
    int s = (e < E) ? ei[e]     : (e - E);
    int d = (e < E) ? ei[E + e] : (e - E);
    float w0 = ex[(size_t)e * 4 + 0] / den[(size_t)d * 4 + 0] * 0.25f;
    float w1 = ex[(size_t)e * 4 + 1] / den[(size_t)d * 4 + 1] * 0.25f;
    float w2 = ex[(size_t)e * 4 + 2] / den[(size_t)d * 4 + 2] * 0.25f;
    float w3 = ex[(size_t)e * 4 + 3] / den[(size_t)d * 4 + 3] * 0.25f;
    const float* pl = xl + (size_t)s * 512;
    float* po = hout + (size_t)d * 128;
    for (int c = 0; c < 128; ++c) {
        float v = pl[c] * w0 + pl[128 + c] * w1 + pl[256 + c] * w2 + pl[384 + c] * w3;
        atomicAdd(&po[c], v);
    }
}

__global__ void gat_post_k(float* __restrict__ h, const float* __restrict__ b,
                           int N, int do_elu) {
    int i = blockIdx.x, t = threadIdx.x;    // grid N, block 128
    if (i >= N) return;
    float v = h[(size_t)i * 128 + t] + b[t];
    if (do_elu) v = (v > 0.0f) ? v : (expf(v) - 1.0f);
    h[(size_t)i * 128 + t] = v;
}

// ---------- graph mean pool ----------
__global__ void pool_k(const float* __restrict__ h, const int* __restrict__ batch,
                       float* __restrict__ dsum, float* __restrict__ cnt, int N) {
    int i = blockIdx.x, t = threadIdx.x;    // grid N, block 128
    if (i >= N) return;
    int b = batch[i];
    atomicAdd(&dsum[(size_t)b * 128 + t], h[(size_t)i * 128 + t]);
    if (t == 0) atomicAdd(&cnt[b], 1.0f);
}

__global__ void pooldiv_k(const float* __restrict__ dsum, const float* __restrict__ cnt,
                          float* __restrict__ demb) {
    int b = blockIdx.x, t = threadIdx.x;
    demb[(size_t)b * 128 + t] = dsum[(size_t)b * 128 + t] / fmaxf(cnt[b], 1.0f);
}

// ---------- embedding gather ----------
__global__ void embed_k(const int* __restrict__ tgt, const float* __restrict__ emb,
                        float* __restrict__ out) {
    int i = blockIdx.x, t = threadIdx.x;    // grid TOK, block 128
    int idx = tgt[i];
    out[(size_t)i * 128 + t] = emb[(size_t)idx * 128 + t];
}

// ---------- fused attention: one block per (b,h), L=512, hd=32 ----------
__global__ void attn_k(const float* __restrict__ qkv, float* __restrict__ out) {
    const int bh = blockIdx.x;
    const int b = bh >> 2, h = bh & 3;
    const int t = threadIdx.x;              // block = 128
    __shared__ float s[512];
    __shared__ float red[128];
    __shared__ float qv[32];
    const float scale = 0.17677669529663687f;  // 1/sqrt(32)
    const float* base = qkv + (size_t)b * 512 * 384;
    for (int lq = 0; lq < 512; ++lq) {
        if (t < 32) qv[t] = base[(size_t)lq * 384 + h * 32 + t];
        __syncthreads();
        for (int lk = t; lk < 512; lk += 128) {
            const float* kk = base + (size_t)lk * 384 + 128 + h * 32;
            float d = 0.0f;
            for (int c = 0; c < 32; ++c) d += qv[c] * kk[c];
            s[lk] = d * scale;
        }
        __syncthreads();
        float m = -1e30f;
        for (int lk = t; lk < 512; lk += 128) m = fmaxf(m, s[lk]);
        red[t] = m; __syncthreads();
        for (int o = 64; o > 0; o >>= 1) { if (t < o) red[t] = fmaxf(red[t], red[t + o]); __syncthreads(); }
        m = red[0]; __syncthreads();
        float sum = 0.0f;
        for (int lk = t; lk < 512; lk += 128) { float e = expf(s[lk] - m); s[lk] = e; sum += e; }
        red[t] = sum; __syncthreads();
        for (int o = 64; o > 0; o >>= 1) { if (t < o) red[t] += red[t + o]; __syncthreads(); }
        float inv = 1.0f / red[0];
        __syncthreads();
        int d = t & 31, ch = t >> 5;
        float acc = 0.0f;
        for (int lk = ch * 128; lk < ch * 128 + 128; ++lk)
            acc += s[lk] * base[(size_t)lk * 384 + 256 + h * 32 + d];
        red[t] = acc * inv; __syncthreads();
        if (t < 32)
            out[((size_t)(b * 512 + lq)) * 128 + h * 32 + t] =
                red[t] + red[t + 32] + red[t + 64] + red[t + 96];
        __syncthreads();
    }
}

// ---------- residual + LayerNorm (D=128), one row per block ----------
__global__ void add_ln_k(const float* __restrict__ x, const float* __restrict__ y,
                         const float* __restrict__ g, const float* __restrict__ bb,
                         float* __restrict__ out) {
    int row = blockIdx.x, t = threadIdx.x;  // block = 128
    __shared__ float red[128];
    float v = x[(size_t)row * 128 + t] + y[(size_t)row * 128 + t];
    red[t] = v; __syncthreads();
    for (int o = 64; o > 0; o >>= 1) { if (t < o) red[t] += red[t + o]; __syncthreads(); }
    float mu = red[0] * (1.0f / 128.0f); __syncthreads();
    float d = v - mu;
    red[t] = d * d; __syncthreads();
    for (int o = 64; o > 0; o >>= 1) { if (t < o) red[t] += red[t + o]; __syncthreads(); }
    float var = red[0] * (1.0f / 128.0f);
    out[(size_t)row * 128 + t] = d * rsqrtf(var + 1e-5f) * g[t] + bb[t];
}

// ---------- mean over sequence (L=512) ----------
__global__ void mean_l_k(const float* __restrict__ pr, float* __restrict__ pm) {
    int b = blockIdx.x, t = threadIdx.x;    // grid B, block 128
    float s = 0.0f;
    for (int l = 0; l < 512; ++l) s += pr[((size_t)b * 512 + l) * 128 + t];
    pm[(size_t)b * 128 + t] = s * (1.0f / 512.0f);
}

// ---------- concat [d_emb | p_emb] ----------
__global__ void concat_k(const float* __restrict__ a, const float* __restrict__ b,
                         float* __restrict__ z) {
    int row = blockIdx.x, t = threadIdx.x;  // grid B, block 256
    z[(size_t)row * 256 + t] = (t < 128) ? a[(size_t)row * 128 + t]
                                         : b[(size_t)row * 128 + (t - 128)];
}

// ---------- host side ----------
static inline void gemm(const float* A, const float* W, const float* b, float* C,
                        int M, int K, int N, int act, hipStream_t s) {
    dim3 g((M + BM - 1) / BM, (N + BN - 1) / BN);
    gemm_tile<<<g, 256, 0, s>>>(A, W, b, C, M, K, N, act);
}
static inline void fill(float* p, float v, size_t n, hipStream_t s) {
    fill_k<<<(unsigned)((n + 255) / 256), 256, 0, s>>>(p, v, n);
}

extern "C" void kernel_launch(void* const* d_in, const int* in_sizes, int n_in,
                              void* d_out, int out_size, void* d_ws, size_t ws_size,
                              hipStream_t stream) {
    const float* x     = (const float*)d_in[0];
    const int*   ei    = (const int*)d_in[1];
    const int*   batch = (const int*)d_in[2];
    const int*   tgt   = (const int*)d_in[3];
    const float* P[64];
    for (int i = 4; i < n_in && i < 64; ++i) P[i] = (const float*)d_in[i];

    const int Nn  = in_sizes[0] / 78;       // 50000
    const int E   = in_sizes[1] / 2;        // 200000
    const int E2  = E + Nn;
    const int TOK = in_sizes[3];            // 65536 = B*L
    const int Bq  = TOK / 512;              // 128
    const size_t CH = 8192;                 // FF row chunk

    float* ws = (float*)d_ws;
    // --- GAT phase view of big region ---
    float* g_h0 = ws;
    float* g_h1 = g_h0 + (size_t)Nn * 128;
    float* g_xl = g_h1 + (size_t)Nn * 128;
    float* g_xr = g_xl + (size_t)Nn * 512;
    float* g_al = g_xr + (size_t)Nn * 512;
    float* g_mx = g_al + (size_t)E2 * 4;
    float* g_dn = g_mx + (size_t)Nn * 4;
    float* g_xp = g_dn + (size_t)Nn * 4;     // x padded to K=96
    float* g_wl = g_xp + (size_t)Nn * 96;    // g0_wl padded
    float* g_wr = g_wl + (size_t)512 * 96;   // g0_wr padded
    size_t gat_end = (size_t)(g_wr + (size_t)512 * 96 - ws);
    // --- transformer phase view (same region, later in time) ---
    float* t_x   = ws;
    float* t_y   = t_x   + (size_t)TOK * 128;
    float* t_qkv = t_y   + (size_t)TOK * 128;
    float* t_at  = t_qkv + (size_t)TOK * 384;
    float* t_ff  = t_at  + (size_t)TOK * 128;
    size_t tr_end = (size_t)(t_ff + CH * 2048 - ws);
    size_t big = gat_end > tr_end ? gat_end : tr_end;
    // --- small persistent buffers ---
    float* sm = ws + big;
    float* dsum  = sm; sm += 128 * 128;
    float* cnt   = sm; sm += 128;
    float* demb  = sm; sm += 128 * 128;
    float* pmean = sm; sm += 128 * 128;
    float* pemb  = sm; sm += 128 * 128;
    float* zc    = sm; sm += 128 * 256;
    float* z1    = sm; sm += 128 * 512;
    float* z2    = sm; sm += 128 * 128;
    float* zd1   = sm; sm += 128 * 64;
    float* zp1   = sm; sm += 128 * 64;

    const int egrid = (E2 + 255) / 256;

    // ================= GATv2 x3 =================
    // pad layer-0 operands to K=96 so the fast GEMM path applies everywhere
    pad_rows_k<<<Nn, 128, 0, stream>>>(x, g_xp, 78, 96);
    pad_rows_k<<<512, 128, 0, stream>>>(P[4], g_wl, 78, 96);
    pad_rows_k<<<512, 128, 0, stream>>>(P[6], g_wr, 78, 96);

    const float* gin = g_xp;
    float* gout = g_h0;
    int curK = 96;
    for (int l = 0; l < 3; ++l) {
        int base = 4 + 6 * l;
        const float* wl = (l == 0) ? g_wl : P[base + 0];
        const float* wr = (l == 0) ? g_wr : P[base + 2];
        gemm(gin, wl, P[base + 1], g_xl, Nn, curK, 512, 0, stream);   // lin_l
        gemm(gin, wr, P[base + 3], g_xr, Nn, curK, 512, 0, stream);   // lin_r
        fill(g_mx, -1e30f, (size_t)Nn * 4, stream);
        fill(g_dn, 0.0f, (size_t)Nn * 4, stream);
        fill(gout, 0.0f, (size_t)Nn * 128, stream);
        gat_alpha_k <<<egrid, 256, 0, stream>>>(ei, g_xl, g_xr, P[base + 4], g_al, E, Nn);
        gat_amax_k  <<<egrid, 256, 0, stream>>>(ei, g_al, g_mx, E, Nn);
        gat_expden_k<<<egrid, 256, 0, stream>>>(ei, g_al, g_mx, g_dn, E, Nn);
        gat_scatter_k<<<egrid, 256, 0, stream>>>(ei, g_al, g_dn, g_xl, gout, E, Nn);
        gat_post_k  <<<Nn, 128, 0, stream>>>(gout, P[base + 5], Nn, (l < 2) ? 1 : 0);
        gin = gout;
        gout = (gout == g_h0) ? g_h1 : g_h0;
        curK = 128;
    }
    const float* hfin = gin;

    // ================= mean pool -> d_emb =================
    fill(dsum, 0.0f, 128 * 128, stream);
    fill(cnt, 0.0f, 128, stream);
    pool_k   <<<Nn, 128, 0, stream>>>(hfin, batch, dsum, cnt, Nn);
    pooldiv_k<<<128, 128, 0, stream>>>(dsum, cnt, demb);

    // ================= transformer =================
    embed_k<<<TOK, 128, 0, stream>>>(tgt, P[46], t_x);
    for (int l = 0; l < 2; ++l) {
        int b = 22 + 12 * l;
        gemm(t_x, P[b + 0], P[b + 1], t_qkv, TOK, 128, 384, 0, stream);        // QKV
        attn_k<<<Bq * 4, 128, 0, stream>>>(t_qkv, t_at);                       // SDPA
        gemm(t_at, P[b + 2], P[b + 3], t_qkv, TOK, 128, 128, 0, stream);       // O (qkv dead)
        add_ln_k<<<TOK, 128, 0, stream>>>(t_x, t_qkv, P[b + 8], P[b + 9], t_y);
        for (size_t r0 = 0; r0 < (size_t)TOK; r0 += CH) {                      // FFN chunks
            gemm(t_y + r0 * 128, P[b + 4], P[b + 5], t_ff, (int)CH, 128, 2048, 1, stream);
            gemm(t_ff, P[b + 6], P[b + 7], t_at + r0 * 128, (int)CH, 2048, 128, 0, stream);
        }
        add_ln_k<<<TOK, 128, 0, stream>>>(t_y, t_at, P[b + 10], P[b + 11], t_x);
    }
    mean_l_k<<<Bq, 128, 0, stream>>>(t_x, pmean);
    gemm(pmean, P[47], P[48], pemb, Bq, 128, 128, 0, stream);                  // fc

    // ================= heads =================
    float* out = (float*)d_out;
    concat_k<<<Bq, 256, 0, stream>>>(demb, pemb, zc);
    gemm(zc, P[49], P[50], z1, Bq, 256, 512, 1, stream);                       // r1 + relu
    gemm(z1, P[51], P[52], z2, Bq, 512, 128, 1, stream);                       // r2 + relu
    dot_head_k<<<Bq, 128, 0, stream>>>(z2, P[53], P[54], out, 128);            // pred (N=1)
    gemm(demb, P[55], P[56], zd1, Bq, 128, 64, 1, stream);                     // dp1 + relu
    gemm(zd1, P[57], P[58], out + 128, Bq, 64, 64, 0, stream);                 // z_d
    gemm(pemb, P[59], P[60], zp1, Bq, 128, 64, 1, stream);                     // pp1 + relu
    gemm(zp1, P[61], P[62], out + 128 + 128 * 64, Bq, 64, 64, 0, stream);      // z_p
}
